// GTN4o_40132174414152
// MI455X (gfx1250) — matrix operand order repaired and checked by reading, not compiled, not run on hardware
//
#include <hip/hip_runtime.h>
#include <hip/hip_bf16.h>
#include <stdint.h>

typedef __attribute__((ext_vector_type(16))) _Float16 v16h;
typedef __attribute__((ext_vector_type(8)))  float    v8f;

#define NN   4096
#define DD   256
#define HH   8
#define DHH  32
#define EE   131072
#define EPP  65536

__device__ __forceinline__ v8f wmma_f16(v16h a, v16h b, v8f c) {
  // 8 args: (neg_a, A, neg_b, B, c_mod, C, reuse_a, reuse_b)
  return __builtin_amdgcn_wmma_f32_16x16x32_f16(false, a, false, b, (short)0, c, false, false);
}

// ---------------------------------------------------------------------------
// Generic WMMA GEMM: C[M,Nout] = A[M,K] @ W[Nout,K]^T + bias[Nout]
// block = 256 threads (8 waves as 4(M) x 2(N)); block tile 64x64;
// wave tile 16x32 (two accumulators sharing one A-fragment -> 2 WMMA / K-step)
// ---------------------------------------------------------------------------
__global__ void k_gemm(const float* __restrict__ A, const float* __restrict__ W,
                       const float* __restrict__ bias, float* __restrict__ C,
                       int K, int Nout) {
  const int wave = threadIdx.x >> 5;
  const int lane = threadIdx.x & 31;
  const int half = lane >> 4;
  const int ln15 = lane & 15;
  const int row0 = blockIdx.y * 64 + (wave >> 1) * 16;
  const int n0   = blockIdx.x * 64 + (wave & 1) * 32;

  v8f acc0 = {}, acc1 = {};
  const float* rA  = A + (row0 + ln15) * K;        // A-frag: lane = M row
  const float* rW0 = W + (n0 + ln15) * K;          // B-frag tile 0: lane = N col
  const float* rW1 = W + (n0 + 16 + ln15) * K;     // B-frag tile 1

  for (int k0 = 0; k0 < K; k0 += 32) {
    __builtin_prefetch(rA + k0 + 32, 0, 0);
    v16h af, bf0, bf1;
#pragma unroll
    for (int v = 0; v < 8; ++v) {
      // A 16x32 f16 layout: v<4 -> K = 2v + 8*half ; v>=4 -> K = 2v + 8 + 8*half
      int ka = k0 + ((v < 4) ? (2 * v + 8 * half) : (2 * v + 8 + 8 * half));
      af[2 * v]     = (_Float16)rA[ka];
      af[2 * v + 1] = (_Float16)rA[ka + 1];
      // B 32x16 f16 layout: same K order as A: v<4 -> K = 2v + 8*half ; v>=4 -> K = 2v + 8 + 8*half
      int kb = k0 + ((v < 4) ? (2 * v + 8 * half) : (2 * v + 8 + 8 * half));
      bf0[2 * v]     = (_Float16)rW0[kb];
      bf0[2 * v + 1] = (_Float16)rW0[kb + 1];
      bf1[2 * v]     = (_Float16)rW1[kb];
      bf1[2 * v + 1] = (_Float16)rW1[kb + 1];
    }
    acc0 = wmma_f16(af, bf0, acc0);
    acc1 = wmma_f16(af, bf1, acc1);
  }

  const int na = n0 + ln15;
  const int nb = n0 + 16 + ln15;
  const float bva = bias[na];
  const float bvb = bias[nb];
#pragma unroll
  for (int v = 0; v < 8; ++v) {
    int m = row0 + v + 8 * half;        // C layout: VGPR v -> M = v + 8*half
    C[m * Nout + na] = acc0[v] + bva;
    C[m * Nout + nb] = acc1[v] + bvb;
  }
}

// ---------------------------------------------------------------------------
// Dense attention bias construction
// ---------------------------------------------------------------------------
__global__ void k_fill_bias(float4* __restrict__ bias) {
  int i = blockIdx.x * blockDim.x + threadIdx.x;
  bias[i] = make_float4(-1e9f, -1e9f, -1e9f, -1e9f);
}

__global__ void k_diag(float* __restrict__ bias) {
  int i = blockIdx.x * blockDim.x + threadIdx.x;
  bias[i * NN + i] = 0.f;
}

__global__ void k_scatter(const int* __restrict__ ei, const float* __restrict__ ew,
                          float* __restrict__ bias) {
  int e = blockIdx.x * blockDim.x + threadIdx.x;
  int src = ei[e];
  int dst = ei[EE + e];
  bias[dst * NN + src] = ew[e];
}

// ---------------------------------------------------------------------------
// Flash attention with dense additive bias.
// grid = (N/64, H), block = 128 (4 waves; each wave owns one 16-query tile).
// Bias tiles are double-buffered into LDS via async global->LDS DMA
// (global_load_async_to_lds_b128, ASYNCcnt), overlapping the next tile's
// transfer with the current chunk's WMMA work.
// Per 32-key chunk: S = Q K^T (2 WMMA) + bias, online softmax, P staged
// through LDS (C-layout -> A-layout), O += P V (2 WMMA).
// ---------------------------------------------------------------------------
__global__ void k_attn(const float* __restrict__ qkv, const float* __restrict__ bias,
                       float* __restrict__ o_out) {
  __shared__ _Float16 sP[4][16][40];      // pad to 40 halfs: pairs stay 4B aligned
  __shared__ float    sB[4][2][16][32];   // per-wave double-buffered bias tiles
  const int wave = threadIdx.x >> 5;
  const int lane = threadIdx.x & 31;
  const int half = lane >> 4;
  const int ln15 = lane & 15;
  const int head = blockIdx.y;
  const int q0 = (blockIdx.x * 4 + wave) * 16;
  const float scale = 0.17677669529663687f;   // 1/sqrt(32)
  const int LD = 3 * DD;                       // qkv row stride

  const float* Qb = qkv + head * DHH;
  const float* Kb = qkv + DD + head * DHH;
  const float* Vb = qkv + 2 * DD + head * DHH;

  // async DMA of one 16x32 f32 bias tile into this wave's LDS buffer:
  // 4 issues; each lane moves 16B (issue j covers rows 4j..4j+3)
  const int brow = lane >> 3;          // 0..3 within issue group
  const int bcg  = (lane & 7) * 4;     // column group (4 floats)
  auto issue_bias_tile = [&](int kbase, int buf) {
#pragma unroll
    for (int j = 0; j < 4; ++j) {
      int row = 4 * j + brow;
      unsigned ldsa = (unsigned)(uintptr_t)&sB[wave][buf][row][bcg];
      uint64_t ga = (uint64_t)(uintptr_t)(bias + (q0 + row) * NN + kbase + bcg);
      asm volatile("global_load_async_to_lds_b128 %0, %1, off"
                   :: "v"(ldsa), "v"(ga) : "memory");
    }
  };

  // Q A-fragment (16 queries x dh=32), loaded once
  v16h qf;
  {
    const float* r = Qb + (q0 + ln15) * LD;
#pragma unroll
    for (int v = 0; v < 8; ++v) {
      int k = (v < 4) ? (2 * v + 8 * half) : (2 * v + 8 + 8 * half);
      qf[2 * v]     = (_Float16)r[k];
      qf[2 * v + 1] = (_Float16)r[k + 1];
    }
  }

  v8f o0 = {}, o1 = {};
  float rmax[8], rsum[8];
#pragma unroll
  for (int v = 0; v < 8; ++v) { rmax[v] = -3.0e38f; rsum[v] = 0.f; }

  issue_bias_tile(0, 0);

  for (int kb = 0; kb < NN; kb += 32) {
    const int buf = (kb >> 5) & 1;
    const bool more = (kb + 32) < NN;
    if (more) issue_bias_tile(kb + 32, buf ^ 1);

    // K B-fragments for two 16-key halves of the chunk
    v16h kf0, kf1;
    {
      const float* r0 = Kb + (kb + ln15) * LD;
      const float* r1 = Kb + (kb + 16 + ln15) * LD;
#pragma unroll
      for (int v = 0; v < 8; ++v) {
        int k = (v < 4) ? (2 * v + 8 * half) : (2 * v + 8 + 8 * half);
        kf0[2 * v] = (_Float16)r0[k];  kf0[2 * v + 1] = (_Float16)r0[k + 1];
        kf1[2 * v] = (_Float16)r1[k];  kf1[2 * v + 1] = (_Float16)r1[k + 1];
      }
    }
    v8f s0 = {}, s1 = {};
    s0 = wmma_f16(qf, kf0, s0);
    s1 = wmma_f16(qf, kf1, s1);

    // wait for current bias tile (keep next tile's 4 loads in flight)
    if (more) asm volatile("s_wait_asynccnt 4" ::: "memory");
    else      asm volatile("s_wait_asynccnt 0" ::: "memory");

    float p0[8], p1[8];
#pragma unroll
    for (int v = 0; v < 8; ++v) {
      int lrow = v + 8 * half;
      p0[v] = s0[v] * scale + sB[wave][buf][lrow][ln15];
      p1[v] = s1[v] * scale + sB[wave][buf][lrow][16 + ln15];
    }
    // online softmax; row reductions stay inside 16-lane half-groups (xor 1..8)
#pragma unroll
    for (int v = 0; v < 8; ++v) {
      float cmax = fmaxf(p0[v], p1[v]);
#pragma unroll
      for (int m = 8; m >= 1; m >>= 1) cmax = fmaxf(cmax, __shfl_xor(cmax, m, 32));
      float nmax = fmaxf(rmax[v], cmax);
      float alpha = __expf(rmax[v] - nmax);
      rmax[v] = nmax;
      p0[v] = __expf(p0[v] - nmax);
      p1[v] = __expf(p1[v] - nmax);
      float csum = p0[v] + p1[v];
#pragma unroll
      for (int m = 8; m >= 1; m >>= 1) csum += __shfl_xor(csum, m, 32);
      rsum[v] = rsum[v] * alpha + csum;
      o0[v] *= alpha;
      o1[v] *= alpha;
    }
    // stage P: C-layout (lane=col) -> LDS -> A-layout (lane=row)
#pragma unroll
    for (int v = 0; v < 8; ++v) {
      sP[wave][v + 8 * half][ln15]      = (_Float16)p0[v];
      sP[wave][v + 8 * half][16 + ln15] = (_Float16)p1[v];
    }
    __syncthreads();
    v16h pf;
#pragma unroll
    for (int v = 0; v < 8; ++v) {
      int k = (v < 4) ? (2 * v + 8 * half) : (2 * v + 8 + 8 * half);
      pf[2 * v]     = sP[wave][ln15][k];
      pf[2 * v + 1] = sP[wave][ln15][k + 1];
    }
    __syncthreads();
    // V B-fragments (32 keys x 16 dh-cols each)
    v16h vf0, vf1;
#pragma unroll
    for (int v = 0; v < 8; ++v) {
      int key = kb + ((v < 4) ? (2 * v + 8 * half) : (2 * v + 8 + 8 * half));
      vf0[2 * v]     = (_Float16)Vb[key * LD + ln15];
      vf0[2 * v + 1] = (_Float16)Vb[(key + 1) * LD + ln15];
      vf1[2 * v]     = (_Float16)Vb[key * LD + 16 + ln15];
      vf1[2 * v + 1] = (_Float16)Vb[(key + 1) * LD + 16 + ln15];
    }
    o0 = wmma_f16(pf, vf0, o0);
    o1 = wmma_f16(pf, vf1, o1);
  }

#pragma unroll
  for (int v = 0; v < 8; ++v) {
    int qrow = q0 + v + 8 * half;
    float inv = 1.0f / rsum[v];
    o_out[qrow * DD + head * DHH + ln15]      = o0[v] * inv;
    o_out[qrow * DD + head * DHH + 16 + ln15] = o1[v] * inv;
  }
}

// ---------------------------------------------------------------------------
// out = LayerNorm(a + b) * g + beta ; one wave per 256-wide row
// ---------------------------------------------------------------------------
__global__ void k_add_ln(const float* __restrict__ a, const float* __restrict__ b,
                         const float* __restrict__ g, const float* __restrict__ be,
                         float* __restrict__ out) {
  int row = blockIdx.x * 8 + (threadIdx.x >> 5);
  int lane = threadIdx.x & 31;
  const float* pa = a + row * DD + lane * 8;
  const float* pb = b + row * DD + lane * 8;
  float x[8];
  float s = 0.f, s2 = 0.f;
#pragma unroll
  for (int i = 0; i < 8; ++i) {
    x[i] = pa[i] + pb[i];
    s += x[i];
    s2 += x[i] * x[i];
  }
#pragma unroll
  for (int m = 16; m >= 1; m >>= 1) {
    s  += __shfl_xor(s, m, 32);
    s2 += __shfl_xor(s2, m, 32);
  }
  float mean = s * (1.0f / DD);
  float var = s2 * (1.0f / DD) - mean * mean;
  float rstd = rsqrtf(var + 1e-5f);
  float* po = out + row * DD + lane * 8;
  const float* pg = g + lane * 8;
  const float* pq = be + lane * 8;
#pragma unroll
  for (int i = 0; i < 8; ++i) po[i] = (x[i] - mean) * rstd * pg[i] + pq[i];
}

__global__ void k_add(const float* __restrict__ a, const float* __restrict__ b,
                      float* __restrict__ out) {
  int i = blockIdx.x * blockDim.x + threadIdx.x;
  out[i] = a[i] + b[i];
}

// ---------------------------------------------------------------------------
// Edge scorer: one wave per pair. z = hc[src]*hc[dst] (lane-distributed 8/lane),
// 128 reduced dots + LeakyReLU(0.2) + final dot with w2.
// ---------------------------------------------------------------------------
__global__ void k_score(const float* __restrict__ hc,
                        const int* __restrict__ src, const int* __restrict__ dst,
                        const float* __restrict__ w1, const float* __restrict__ b1,
                        const float* __restrict__ w2, const float* __restrict__ b2,
                        float* __restrict__ out) {
  int pair = blockIdx.x * 8 + (threadIdx.x >> 5);
  int lane = threadIdx.x & 31;
  int si = src[pair], di = dst[pair];
  const float* ps = hc + si * DD + lane * 8;
  const float* pd = hc + di * DD + lane * 8;
  float z[8];
#pragma unroll
  for (int i = 0; i < 8; ++i) z[i] = ps[i] * pd[i];
  float o = b2[0];
  for (int j = 0; j < 128; ++j) {
    const float* wr = w1 + j * DD + lane * 8;
    float p = 0.f;
#pragma unroll
    for (int i = 0; i < 8; ++i) p += z[i] * wr[i];
#pragma unroll
    for (int m = 16; m >= 1; m >>= 1) p += __shfl_xor(p, m, 32);
    p += b1[j];
    p = (p >= 0.f) ? p : 0.2f * p;
    o += p * w2[j];
  }
  if (lane == 0) out[pair] = o;
}

// ---------------------------------------------------------------------------
extern "C" void kernel_launch(void* const* d_in, const int* in_sizes, int n_in,
                              void* d_out, int out_size, void* d_ws, size_t ws_size,
                              hipStream_t stream) {
  (void)in_sizes; (void)n_in; (void)out_size; (void)ws_size;

  const float* x   = (const float*)d_in[0];
  const int*   ei[2] = { (const int*)d_in[1], (const int*)d_in[3] };
  const float* ew[2] = { (const float*)d_in[2], (const float*)d_in[4] };
  const int* pos_src = (const int*)d_in[5];
  const int* pos_dst = (const int*)d_in[6];
  const int* neg_src = (const int*)d_in[7];
  const int* neg_dst = (const int*)d_in[8];
  const float* ip_w = (const float*)d_in[9];
  const float* ip_b = (const float*)d_in[10];
  const float* pw1 = (const float*)d_in[31];
  const float* pb1 = (const float*)d_in[32];
  const float* pw2 = (const float*)d_in[33];
  const float* pb2 = (const float*)d_in[34];

  float* ws = (float*)d_ws;
  float* h_proj = ws;                       // 4096*256
  float* h      = ws + 1048576;             // 4096*256
  float* qkv    = ws + 2097152;             // 4096*768
  float* o      = ws + 5242880;             // 4096*256
  float* tmp    = ws + 6291456;             // 4096*256
  float* bias   = ws + 7340032;             // 4096*4096

  float* out_pos = (float*)d_out;
  float* out_neg = out_pos + EPP;
  float* hc      = out_neg + EPP;           // 4096*256

  // h_proj = x @ ip_w^T + ip_b           (4096 x 128 -> 256)
  k_gemm<<<dim3(DD / 64, NN / 64), 256, 0, stream>>>(x, ip_w, ip_b, h_proj, 128, DD);

  const float* h_in = h_proj;
  for (int l = 0; l < 2; ++l) {
    const float* in_w  = (const float*)d_in[11 + 10 * l + 0];
    const float* in_b  = (const float*)d_in[11 + 10 * l + 1];
    const float* out_w = (const float*)d_in[11 + 10 * l + 2];
    const float* out_b = (const float*)d_in[11 + 10 * l + 3];
    const float* ff_w  = (const float*)d_in[11 + 10 * l + 4];
    const float* ff_b  = (const float*)d_in[11 + 10 * l + 5];
    const float* n1_g  = (const float*)d_in[11 + 10 * l + 6];
    const float* n1_b  = (const float*)d_in[11 + 10 * l + 7];
    const float* n2_g  = (const float*)d_in[11 + 10 * l + 8];
    const float* n2_b  = (const float*)d_in[11 + 10 * l + 9];

    // dense additive bias (L2-resident 64 MB)
    k_fill_bias<<<(NN * (NN / 4)) / 256, 256, 0, stream>>>((float4*)bias);
    k_diag<<<NN / 256, 256, 0, stream>>>(bias);
    k_scatter<<<EE / 256, 256, 0, stream>>>(ei[l], ew[l], bias);

    // qkv = h_in @ in_w^T + in_b          (4096 x 256 -> 768)
    k_gemm<<<dim3((3 * DD) / 64, NN / 64), 256, 0, stream>>>(h_in, in_w, in_b, qkv, DD, 3 * DD);

    // flash attention with bias -> o      (4096 x 256)
    k_attn<<<dim3(NN / 64, HH), 128, 0, stream>>>(qkv, bias, o);

    // tmp = o @ out_w^T + out_b
    k_gemm<<<dim3(DD / 64, NN / 64), 256, 0, stream>>>(o, out_w, out_b, tmp, DD, DD);
    // h = LN(h_in + tmp)
    k_add_ln<<<NN / 8, 256, 0, stream>>>(h_in, tmp, n1_g, n1_b, h);
    // tmp = h @ ff_w^T + ff_b
    k_gemm<<<dim3(DD / 64, NN / 64), 256, 0, stream>>>(h, ff_w, ff_b, tmp, DD, DD);
    // h = LN(h + tmp)
    k_add_ln<<<NN / 8, 256, 0, stream>>>(h, tmp, n2_g, n2_b, h);

    h_in = h;
  }

  // h_combined = h + h_proj  (third output)
  k_add<<<(NN * DD) / 256, 256, 0, stream>>>(h, h_proj, hc);

  // scores
  k_score<<<EPP / 8, 256, 0, stream>>>(hc, pos_src, pos_dst, pw1, pb1, pw2, pb2, out_pos);
  k_score<<<EPP / 8, 256, 0, stream>>>(hc, neg_src, neg_dst, pw1, pb1, pw2, pb2, out_neg);
}
